// CrossAttention_91122026152638
// MI455X (gfx1250) — matrix operand order repaired and checked by reading, not compiled, not run on hardware
//
#include <hip/hip_runtime.h>
#include <hip/hip_bf16.h>
#include <math.h>

typedef __attribute__((ext_vector_type(16))) _Float16 v16h;
typedef __attribute__((ext_vector_type(8)))  _Float16 v8h;
typedef __attribute__((ext_vector_type(8)))  float    v8f;

#define HEADS   8
#define DIMH    40
#define DPAD    64
#define SEQ     4096
#define BATCH   2
#define DMODEL  320
#define MROWS   (BATCH*SEQ)   /* 8192 */
#define BC      64            /* key block for flash attention */

// ---- workspace byte offsets ----
#define WS_XH    0           /* 8192*320 f16  = 5,242,880 B */
#define WS_WQKV  5242880     /*  960*320 f16  =   614,400 B */
#define WS_WOUT  5857280     /*  320*320 f16  =   204,800 B */
#define WS_QH    6062080     /* 16*4096*64 f16 = 8,388,608 B */
#define WS_KH    14450688
#define WS_VH    22839296
#define WS_OH    31227904    /* 8192*320 f16  = 5,242,880 B */

// A-matrix 16x32 f16 tile load: lane m in [0,16) holds K{0..7,16..23},
// lane m+16 holds K{8..15,24..31}.  p = base + (lane&15)*ld + ((lane>>4)<<3)
__device__ __forceinline__ v16h ldA16(const _Float16* p) {
  v8h lo = *(const v8h*)p;
  v8h hi = *(const v8h*)(p + 16);
  v16h r;
#pragma unroll
  for (int i = 0; i < 8; ++i) { r[i] = lo[i]; r[i + 8] = hi[i]; }
  return r;
}

// B-matrix 32x16 f16 tile load: lane n in [0,16) holds K{0..7,16..23} of column n,
// lane n+16 holds K{8..15,24..31}.  p = base + (lane&15)*ld + ((lane>>4)<<3)
__device__ __forceinline__ v16h ldB16(const _Float16* p) {
  v8h lo = *(const v8h*)p;
  v8h hi = *(const v8h*)(p + 16);
  v16h r;
#pragma unroll
  for (int i = 0; i < 8; ++i) { r[i] = lo[i]; r[i + 8] = hi[i]; }
  return r;
}

__device__ __forceinline__ v8f wmma_f16(v16h a, v16h b, v8f c) {
  return __builtin_amdgcn_wmma_f32_16x16x32_f16(
      /*neg_a=*/false, a, /*neg_b=*/false, b,
      /*c_mod=*/(short)0, c, /*reuse_a=*/false, /*reuse_b=*/false);
}

__device__ __forceinline__ v8f zero8() {
  v8f z = {0.f, 0.f, 0.f, 0.f, 0.f, 0.f, 0.f, 0.f};
  return z;
}

// Async global->LDS copy, 16 bytes per lane (CDNA5, tracked by ASYNCcnt).
// VDST = LDS byte offset (low 32 bits of the shared-aperture flat address),
// VADDR = 64-bit global address, GV mode (saddr = off).
__device__ __forceinline__ void async_ld_b128(_Float16* lds_dst,
                                              const _Float16* gsrc) {
  unsigned lo = (unsigned)(uintptr_t)lds_dst;
  unsigned long long ga = (unsigned long long)(uintptr_t)gsrc;
  asm volatile("global_load_async_to_lds_b128 %0, %1, off"
               :: "v"(lo), "v"(ga) : "memory");
}

__device__ __forceinline__ void wait_async0() {
  asm volatile("s_wait_asynccnt 0x0" ::: "memory");
}

// ------------------------------------------------------------------ utilities
__global__ void zero_ws_kernel(uint4* p, long n) {
  long i = (long)blockIdx.x * blockDim.x + threadIdx.x;
  if (i < n) p[i] = make_uint4(0u, 0u, 0u, 0u);
}

__global__ void f32_to_f16_kernel(const float* __restrict__ src,
                                  _Float16* __restrict__ dst, int n) {
  int i = blockIdx.x * blockDim.x + threadIdx.x;
  if (i < n) dst[i] = (_Float16)src[i];
}

// ------------------------------------------------------------------ QKV GEMM
// C[8192,960] = xh[8192,320] * Wqkv[960,320]^T, scattered to padded per-head
// f16 q/k/v buffers [B,h,S,64].
__global__ __launch_bounds__(256) void gemm_qkv_kernel(
    const _Float16* __restrict__ A, const _Float16* __restrict__ Bw,
    _Float16* __restrict__ qh, _Float16* __restrict__ kh,
    _Float16* __restrict__ vh) {
  const int lane = threadIdx.x & 31;
  const int wave = threadIdx.x >> 5;
  const int l15  = lane & 15;
  const int aoff = (lane >> 4) << 3;
  const int boff = (lane >> 4) << 3;
  const int rsel = (lane >> 4) << 3;
  const int row0 = blockIdx.x * 128 + wave * 16;
  const int col0 = blockIdx.y * 64;

  v8f acc[4];
#pragma unroll
  for (int t = 0; t < 4; ++t) acc[t] = zero8();

  const _Float16* arow = A + (size_t)(row0 + l15) * DMODEL + aoff;
#pragma unroll
  for (int kc = 0; kc < DMODEL / 32; ++kc) {
    v16h a = ldA16(arow + kc * 32);
#pragma unroll
    for (int t = 0; t < 4; ++t) {
      const _Float16* bp =
          Bw + (size_t)(col0 + t * 16 + l15) * DMODEL + kc * 32 + boff;
      acc[t] = wmma_f16(a, ldB16(bp), acc[t]);
    }
  }

#pragma unroll
  for (int t = 0; t < 4; ++t) {
#pragma unroll
    for (int r = 0; r < 8; ++r) {
      int m = row0 + r + rsel;
      int n = col0 + t * 16 + l15;
      int which = n / DMODEL;       // 0=q 1=k 2=v (tile never spans: 320%16==0)
      int c  = n - which * DMODEL;
      int h  = c / DIMH;
      int dd = c - h * DIMH;
      int b_ = m >> 12;
      int s_ = m & (SEQ - 1);
      _Float16* dst = (which == 0) ? qh : (which == 1) ? kh : vh;
      dst[(((size_t)(b_ * HEADS + h)) * SEQ + s_) * DPAD + dd] =
          (_Float16)acc[t][r];
    }
  }
}

// ------------------------------------------------------------------ attention
// grid: (S/128, B*H). 8 waves/block, each wave owns 16 query rows.
// K/V tiles (64 keys) double-buffered in LDS via async global->LDS copies.
__global__ __launch_bounds__(256) void flash_attn_kernel(
    const _Float16* __restrict__ qh, const _Float16* __restrict__ kh,
    const _Float16* __restrict__ vh, _Float16* __restrict__ oh) {
  __shared__ _Float16 lK[2][BC * DPAD];     // keys row-major, double buffered
  __shared__ _Float16 lVraw[2][BC * DPAD];  // V row-major, double buffered
  __shared__ _Float16 lVt[DPAD * BC];       // V transposed [64][64]
  __shared__ _Float16 lP[8 * 16 * BC];      // per-wave P tile [16][64]

  const int lane = threadIdx.x & 31;
  const int wave = threadIdx.x >> 5;
  const int l15  = lane & 15;
  const int aoff = (lane >> 4) << 3;
  const int boff = (lane >> 4) << 3;
  const int rsel = (lane >> 4) << 3;
  const int bhead = blockIdx.y;
  const size_t base = (size_t)bhead * SEQ * DPAD;
  const int q0 = blockIdx.x * 128 + wave * 16;
  const float SC = 0.15811388300841898f;  // 40^-0.5

  const _Float16* qp = qh + base + (size_t)(q0 + l15) * DPAD + aoff;
  v16h qa0 = ldA16(qp);        // d 0..31
  v16h qa1 = ldA16(qp + 32);   // d 32..63 (pad zeros)

  float mrun[8], lrun[8];
  v8f o[4];
#pragma unroll
  for (int r = 0; r < 8; ++r) { mrun[r] = -3.0e38f; lrun[r] = 0.f; }
#pragma unroll
  for (int t = 0; t < 4; ++t) o[t] = zero8();

  _Float16* lPw = lP + wave * (16 * BC);
  const int rr = threadIdx.x >> 2;        // 0..63 : key row this thread stages
  const int cc = (threadIdx.x & 3) * 16;  // 0,16,32,48 : d chunk

  // prologue: stage tile 0 into buffer 0
  {
    const size_t g = base + (size_t)rr * DPAD + cc;
    async_ld_b128(&lK[0][rr * DPAD + cc],        kh + g);
    async_ld_b128(&lK[0][rr * DPAD + cc + 8],    kh + g + 8);
    async_ld_b128(&lVraw[0][rr * DPAD + cc],     vh + g);
    async_ld_b128(&lVraw[0][rr * DPAD + cc + 8], vh + g + 8);
  }

  for (int kb = 0; kb < SEQ / BC; ++kb) {
    const int cur = kb & 1;
    wait_async0();        // this wave's async copies for tile kb landed
    __syncthreads();      // everyone's landed; prev iteration compute done

    // prefetch tile kb+1 into the other buffer (overlaps with compute below)
    if (kb + 1 < SEQ / BC) {
      const size_t g = base + (size_t)((kb + 1) * BC + rr) * DPAD + cc;
      async_ld_b128(&lK[cur ^ 1][rr * DPAD + cc],        kh + g);
      async_ld_b128(&lK[cur ^ 1][rr * DPAD + cc + 8],    kh + g + 8);
      async_ld_b128(&lVraw[cur ^ 1][rr * DPAD + cc],     vh + g);
      async_ld_b128(&lVraw[cur ^ 1][rr * DPAD + cc + 8], vh + g + 8);
    }

    // transpose V tile: lVraw[cur][key][d] -> lVt[d][key]
    {
      v8h a0 = *(const v8h*)&lVraw[cur][rr * DPAD + cc];
      v8h a1 = *(const v8h*)&lVraw[cur][rr * DPAD + cc + 8];
#pragma unroll
      for (int i = 0; i < 8; ++i) {
        lVt[(cc + i) * BC + rr]     = a0[i];
        lVt[(cc + 8 + i) * BC + rr] = a1[i];
      }
    }
    __syncthreads();

    // S = Q K^T  (four 16x16 j-tiles, K-dim = 64 padded d)
    v8f s[4];
#pragma unroll
    for (int j = 0; j < 4; ++j) {
      s[j] = zero8();
      const _Float16* kp = &lK[cur][(j * 16 + l15) * DPAD + boff];
      s[j] = wmma_f16(qa0, ldB16(kp), s[j]);
      s[j] = wmma_f16(qa1, ldB16(kp + 32), s[j]);
    }

    // online softmax (row r lives in VGPR r across a 16-lane group)
    float mnew[8], scl[8];
#pragma unroll
    for (int r = 0; r < 8; ++r) {
      float v = fmaxf(fmaxf(s[0][r], s[1][r]), fmaxf(s[2][r], s[3][r]));
      v = fmaxf(v, __shfl_xor(v, 1, 32));
      v = fmaxf(v, __shfl_xor(v, 2, 32));
      v = fmaxf(v, __shfl_xor(v, 4, 32));
      v = fmaxf(v, __shfl_xor(v, 8, 32));
      mnew[r] = fmaxf(mrun[r], v * SC);
      scl[r]  = __expf(mrun[r] - mnew[r]);
      mrun[r] = mnew[r];
    }
#pragma unroll
    for (int r = 0; r < 8; ++r) {
      const int mr = r + rsel;
      float rs = 0.f;
#pragma unroll
      for (int j = 0; j < 4; ++j) {
        float p = __expf(s[j][r] * SC - mnew[r]);
        rs += p;
        lPw[mr * BC + j * 16 + l15] = (_Float16)p;
      }
      rs += __shfl_xor(rs, 1, 32);
      rs += __shfl_xor(rs, 2, 32);
      rs += __shfl_xor(rs, 4, 32);
      rs += __shfl_xor(rs, 8, 32);
      lrun[r] = lrun[r] * scl[r] + rs;
    }
    asm volatile("s_wait_dscnt 0" ::: "memory");  // wave-local P store->load
    v16h pa0 = ldA16(lPw + l15 * BC + aoff);        // keys 0..31
    v16h pa1 = ldA16(lPw + l15 * BC + 32 + aoff);   // keys 32..63

    // O = O*scale + P V   (4 d-tiles of 16, 2 key chunks each)
#pragma unroll
    for (int t = 0; t < 4; ++t) {
#pragma unroll
      for (int r = 0; r < 8; ++r) o[t][r] *= scl[r];
      const _Float16* vp = &lVt[(t * 16 + l15) * BC + boff];
      o[t] = wmma_f16(pa0, ldB16(vp), o[t]);
      o[t] = wmma_f16(pa1, ldB16(vp + 32), o[t]);
    }
    // loop-head barrier of next iteration orders lVt/lK reuse
  }

  // epilogue: O/l, strip pad (d<40), store head-interleaved f16 [B*S,320]
  const int b_ = bhead >> 3, h_ = bhead & 7;
#pragma unroll
  for (int t = 0; t < 3; ++t) {   // t=3 covers d 48..63: all pad
#pragma unroll
    for (int r = 0; r < 8; ++r) {
      int col = t * 16 + l15;
      if (col < DIMH) {
        int srow = q0 + r + rsel;
        oh[((size_t)(b_ * SEQ + srow)) * DMODEL + h_ * DIMH + col] =
            (_Float16)(o[t][r] / lrun[r]);
      }
    }
  }
}

// ------------------------------------------------------------------ out GEMM
// out[8192,320] = oh[8192,320] * Wout[320,320]^T + bias   (f32 output)
__global__ __launch_bounds__(256) void gemm_out_kernel(
    const _Float16* __restrict__ A, const _Float16* __restrict__ Bw,
    const float* __restrict__ bias, float* __restrict__ out) {
  const int lane = threadIdx.x & 31;
  const int wave = threadIdx.x >> 5;
  const int l15  = lane & 15;
  const int aoff = (lane >> 4) << 3;
  const int boff = (lane >> 4) << 3;
  const int rsel = (lane >> 4) << 3;
  const int row0 = blockIdx.x * 128 + wave * 16;
  const int col0 = blockIdx.y * 64;

  v8f acc[4];
#pragma unroll
  for (int t = 0; t < 4; ++t) acc[t] = zero8();

  const _Float16* arow = A + (size_t)(row0 + l15) * DMODEL + aoff;
#pragma unroll
  for (int kc = 0; kc < DMODEL / 32; ++kc) {
    v16h a = ldA16(arow + kc * 32);
#pragma unroll
    for (int t = 0; t < 4; ++t) {
      const _Float16* bp =
          Bw + (size_t)(col0 + t * 16 + l15) * DMODEL + kc * 32 + boff;
      acc[t] = wmma_f16(a, ldB16(bp), acc[t]);
    }
  }

#pragma unroll
  for (int t = 0; t < 4; ++t) {
#pragma unroll
    for (int r = 0; r < 8; ++r) {
      int m = row0 + r + rsel;
      int n = col0 + t * 16 + l15;
      out[(size_t)m * DMODEL + n] = acc[t][r] + bias[n];
    }
  }
}

// ------------------------------------------------------------------ launcher
extern "C" void kernel_launch(void* const* d_in, const int* in_sizes, int n_in,
                              void* d_out, int out_size, void* d_ws,
                              size_t ws_size, hipStream_t stream) {
  (void)in_sizes; (void)n_in; (void)out_size; (void)ws_size;
  const float* x    = (const float*)d_in[0];
  const float* Wq   = (const float*)d_in[1];
  const float* Wk   = (const float*)d_in[2];
  const float* Wv   = (const float*)d_in[3];
  const float* Wout = (const float*)d_in[4];
  const float* bout = (const float*)d_in[5];
  float* out = (float*)d_out;

  char* ws = (char*)d_ws;
  _Float16* xh   = (_Float16*)(ws + WS_XH);
  _Float16* wqkv = (_Float16*)(ws + WS_WQKV);
  _Float16* wout = (_Float16*)(ws + WS_WOUT);
  _Float16* qh   = (_Float16*)(ws + WS_QH);
  _Float16* khb  = (_Float16*)(ws + WS_KH);
  _Float16* vhb  = (_Float16*)(ws + WS_VH);
  _Float16* ohh  = (_Float16*)(ws + WS_OH);

  // zero padded q/k/v (contiguous region) so d 40..63 contributes nothing
  {
    long n16 = (3L * 16 * SEQ * DPAD * 2) / 16;   // 1,572,864 uint4
    zero_ws_kernel<<<dim3((unsigned)((n16 + 255) / 256)), dim3(256), 0,
                     stream>>>((uint4*)qh, n16);
  }
  f32_to_f16_kernel<<<dim3(10240), dim3(256), 0, stream>>>(x, xh,
                                                           MROWS * DMODEL);
  f32_to_f16_kernel<<<dim3(400), dim3(256), 0, stream>>>(Wq, wqkv, 320 * 320);
  f32_to_f16_kernel<<<dim3(400), dim3(256), 0, stream>>>(Wk, wqkv + 102400,
                                                         320 * 320);
  f32_to_f16_kernel<<<dim3(400), dim3(256), 0, stream>>>(Wv, wqkv + 204800,
                                                         320 * 320);
  f32_to_f16_kernel<<<dim3(400), dim3(256), 0, stream>>>(Wout, wout,
                                                         320 * 320);

  gemm_qkv_kernel<<<dim3(64, 15), dim3(256), 0, stream>>>(xh, wqkv, qh, khb,
                                                          vhb);
  flash_attn_kernel<<<dim3(32, 16), dim3(256), 0, stream>>>(qh, khb, vhb, ohh);
  gemm_out_kernel<<<dim3(64, 5), dim3(256), 0, stream>>>(ohh, wout, bout, out);
}
